// Transformer_E_23759759081507
// MI455X (gfx1250) — compile-verified, matrix operand order checked by reading, not run on hardware
//
#include <hip/hip_runtime.h>
#include <hip/hip_bf16.h>

#define B_SZ   16
#define S_LEN  1024
#define DIM    256
#define HEADS  4
#define HEAD   64
#define MLP    512
#define BS     (B_SZ * S_LEN)   // 16384 token rows

typedef _Float16 h16;
typedef __attribute__((ext_vector_type(16))) _Float16 v16h;
typedef __attribute__((ext_vector_type(8)))  float    v8f;
typedef __attribute__((ext_vector_type(4)))  unsigned int v4u;
typedef __attribute__((ext_vector_type(8)))  int      v8i;
typedef __attribute__((ext_vector_type(4)))  int      v4i;

// ---------------------------------------------------------------------------
// WMMA wrapper: D = A(16x32 f16) x B(32x16 f16) + C(16x16 f32)
// ---------------------------------------------------------------------------
__device__ __forceinline__ v8f wmma_f16(v16h a, v16h b, v8f c) {
  return __builtin_amdgcn_wmma_f32_16x16x32_f16(
      /*neg_a=*/false, a, /*neg_b=*/false, b,
      /*c_mod=*/(short)0, c, /*reuse_a=*/false, /*reuse_b=*/false);
}

// Load a 16x32 f16 fragment (A layout; B mirrored with operand stored N-major).
// Lane l: row (l&15); lanes 0-15 hold K 0..7 / 16..23, lanes 16-31 hold
// K 8..15 / 24..31 — two contiguous 16B runs -> two b128 loads.
__device__ __forceinline__ v16h load_frag(const h16* base, int ld) {
  const int lane = threadIdx.x & 31;
  const h16* p = base + (size_t)(lane & 15) * ld + ((lane >> 4) << 3);
  v16h f;
#pragma unroll
  for (int v = 0; v < 8; ++v) {
    const int k = ((v >> 2) << 4) + ((v & 3) << 1);  // 0,2,4,6, 16,18,20,22
    f[2 * v]     = p[k];
    f[2 * v + 1] = p[k + 1];
  }
  return f;
}

// ---------------------------------------------------------------------------
// TDM: 2D tile (64 x 64 f16) load Global -> LDS via Tensor Data Mover.
// D# packing per CDNA5 ISA (§8.3/8.4):
//   group0: count=1 | lds_addr | global_addr | type=2
//   group1: data_size=1(2B), tensor_dim0=stride, tensor_dim1=0xFFFF,
//           tile_dim0=64, tile_dim1=64, tensor_dim0_stride=stride
// 6-arg clang-23 builtin form; groups 2/3 zero (2D tensor).
// ---------------------------------------------------------------------------
#if __has_builtin(__builtin_amdgcn_tensor_load_to_lds)
#define HAVE_TDM 1
__device__ __forceinline__ void tdm_load_tile_64x64(
    unsigned int lds_off, const h16* gptr, unsigned int stride_elems) {
  unsigned long long ga = (unsigned long long)(uintptr_t)gptr;
  v4u g0;
  g0[0] = 1u;                                   // count=1 (valid), user mode
  g0[1] = lds_off;                              // lds_addr (bytes)
  g0[2] = (unsigned int)ga;                     // global_addr lo
  g0[3] = (unsigned int)(ga >> 32) | 0x80000000u;  // addr hi | type=2
  v8i g1;
  g1[0] = (int)0x00010000u;                     // data_size=1 (2 bytes)
  g1[1] = (int)((stride_elems & 0xFFFFu) << 16);        // tensor_dim0 lo16
  g1[2] = (int)((stride_elems >> 16) | 0xFFFF0000u);    // dim0 hi | dim1 lo=0xFFFF
  g1[3] = (int)(64u << 16);                     // dim1 hi=0 | tile_dim0=64
  g1[4] = 64;                                   // tile_dim1=64, tile_dim2=0
  g1[5] = (int)stride_elems;                    // tensor_dim0_stride lo32
  g1[6] = 0;
  g1[7] = 0;
  v4i z4 = {0, 0, 0, 0};
  v8i z8 = {0, 0, 0, 0, 0, 0, 0, 0};
  __builtin_amdgcn_tensor_load_to_lds(g0, g1, z4, z4, z8, 0);
}
#else
#define HAVE_TDM 0
#endif

// ---------------------------------------------------------------------------
// Weight convert + transpose:  dst[(h*N+n)*K + k] = (f16) src[(h*K+k)*N + n]
// ---------------------------------------------------------------------------
__global__ void transpose_cvt_kernel(const float* __restrict__ src,
                                     h16* __restrict__ dst,
                                     int K, int N, int H) {
  int idx = blockIdx.x * blockDim.x + threadIdx.x;
  int total = H * K * N;
  if (idx >= total) return;
  int h   = idx / (K * N);
  int rem = idx - h * K * N;
  int k   = rem / N;
  int n   = rem - k * N;
  dst[((size_t)(h * N + n)) * K + k] = (h16)src[idx];
}

// ---------------------------------------------------------------------------
// Row LayerNorm over DIM=256: one 256-thread block per token row.
// ---------------------------------------------------------------------------
__global__ __launch_bounds__(256) void ln_single(const float* __restrict__ src,
                                                 const float* __restrict__ w,
                                                 const float* __restrict__ b,
                                                 h16* __restrict__ dst) {
  const size_t row = blockIdx.x;
  const int t = threadIdx.x;
  float v = src[row * DIM + t];
  float s1 = v, s2 = v * v;
#pragma unroll
  for (int m = 1; m < 32; m <<= 1) {
    s1 += __shfl_xor(s1, m, 32);
    s2 += __shfl_xor(s2, m, 32);
  }
  __shared__ float r1[8], r2[8];
  const int wid = t >> 5, lane = t & 31;
  if (lane == 0) { r1[wid] = s1; r2[wid] = s2; }
  __syncthreads();
  float a1 = r1[lane & 7], a2 = r2[lane & 7];
#pragma unroll
  for (int m = 1; m < 8; m <<= 1) {
    a1 += __shfl_xor(a1, m, 32);
    a2 += __shfl_xor(a2, m, 32);
  }
  const float mu  = a1 * (1.0f / DIM);
  const float var = a2 * (1.0f / DIM) - mu * mu;
  const float rs  = rsqrtf(var + 1e-5f);
  dst[row * DIM + t] = (h16)((v - mu) * rs * w[t] + b[t]);
}

// ---------------------------------------------------------------------------
// Generic WMMA GEMM: C[M,N] = act(A[M,K] * Bt[N,K]^T + bias) + resid
// 128 threads = 4 waves; block tile 64x64; wave: 16 rows x 64 cols.
// Software-pipelined: next k-chunk fragments are in flight during WMMAs.
// Optional transposed f16 store (Cf16t[col*M + row]) for the V projection.
// ---------------------------------------------------------------------------
__global__ __launch_bounds__(128) void gemm_kernel(
    const h16* __restrict__ A, const h16* __restrict__ Bt,
    const float* __restrict__ bias, const float* __restrict__ resid,
    h16* __restrict__ Cf16, h16* __restrict__ Cf16t, float* __restrict__ Cf32,
    int M, int N, int K, int act) {
  const int lane = threadIdx.x & 31;
  const int wave = threadIdx.x >> 5;
  const int row0 = blockIdx.x * 64 + wave * 16;
  const int col0 = blockIdx.y * 64;

  const h16* arow = A + (size_t)row0 * K;
  const h16* b0p  = Bt + (size_t)(col0 +  0) * K;
  const h16* b1p  = Bt + (size_t)(col0 + 16) * K;
  const h16* b2p  = Bt + (size_t)(col0 + 32) * K;
  const h16* b3p  = Bt + (size_t)(col0 + 48) * K;

  v8f acc[4] = {};
  const int nk = K >> 5;
  v16h af  = load_frag(arow, K);
  v16h bf0 = load_frag(b0p, K);
  v16h bf1 = load_frag(b1p, K);
  v16h bf2 = load_frag(b2p, K);
  v16h bf3 = load_frag(b3p, K);

  for (int i = 0; i < nk; ++i) {
    v16h afn = af, bf0n = bf0, bf1n = bf1, bf2n = bf2, bf3n = bf3;
    if (i + 1 < nk) {
      const int kc = (i + 1) << 5;
      afn  = load_frag(arow + kc, K);
      bf0n = load_frag(b0p + kc, K);
      bf1n = load_frag(b1p + kc, K);
      bf2n = load_frag(b2p + kc, K);
      bf3n = load_frag(b3p + kc, K);
    }
    if (i + 2 < nk) {  // L2/WGP$ prefetch two chunks ahead
      __builtin_prefetch((const void*)(arow + ((i + 2) << 5)), 0, 0);
    }
    acc[0] = wmma_f16(af, bf0, acc[0]);
    acc[1] = wmma_f16(af, bf1, acc[1]);
    acc[2] = wmma_f16(af, bf2, acc[2]);
    acc[3] = wmma_f16(af, bf3, acc[3]);
    af = afn; bf0 = bf0n; bf1 = bf1n; bf2 = bf2n; bf3 = bf3n;
  }

#pragma unroll
  for (int t = 0; t < 4; ++t) {
    const int col = col0 + t * 16 + (lane & 15);
    const float bv = bias ? bias[col] : 0.0f;
#pragma unroll
    for (int r = 0; r < 8; ++r) {
      const int row = row0 + r + ((lane >> 4) << 3);
      float v = acc[t][r] + bv;
      if (act == 1) v = (v > 0.0f) ? v : 0.01f * v;   // leaky_relu
      const size_t o = (size_t)row * N + col;
      if (resid) v += resid[o];
      if (Cf32)  Cf32[o] = v;
      if (Cf16)  Cf16[o] = (h16)v;
      if (Cf16t) Cf16t[(size_t)col * M + row] = (h16)v;  // transposed store
    }
  }
}

// ---------------------------------------------------------------------------
// Fused retention core + GroupNorm + swish gate.
// Grid: (S/64 query row-blocks, B*HEADS).  128 threads = 4 waves.
// K tile [key][dim] and V^T tile [dim][key] staged in LDS by the TDM.
// ---------------------------------------------------------------------------
__global__ __launch_bounds__(128) void retention_kernel(
    const h16* __restrict__ Q, const h16* __restrict__ K,
    const h16* __restrict__ Vt, const h16* __restrict__ G,
    const float* __restrict__ gn_w, const float* __restrict__ gn_b,
    h16* __restrict__ P) {
  __shared__ h16 ldsK[64 * 64];   // [key][dim]
  __shared__ h16 ldsVt[64 * 64];  // [dim][key]
  __shared__ h16 ldsS[64 * 64];   // [row][key] scaled scores staging

  const int rb   = blockIdx.x;
  const int bh   = blockIdx.y;
  const int b    = bh / HEADS;
  const int hh   = bh % HEADS;
  const int lane = threadIdx.x & 31;
  const int wave = threadIdx.x >> 5;

  // RetNet gamma: 1 - exp(linspace(ln 1/32, ln 1/512, H))
  const float lg0   = logf(1.0f / 32.0f);
  const float step  = (logf(1.0f / 512.0f) - lg0) / (float)(HEADS - 1);
  const float gamma = 1.0f - expf(lg0 + (float)hh * step);
  const float log2g = log2f(gamma);

  const int row0 = rb * 64 + wave * 16;
  const h16* qbase = Q + ((size_t)b * S_LEN + row0) * DIM + hh * HEAD;
  const v16h aq0 = load_frag(qbase + 0,  DIM);
  const v16h aq1 = load_frag(qbase + 32, DIM);

  v8f acc[4] = {};

  for (int kb = 0; kb <= rb; ++kb) {
    __syncthreads();  // previous iteration's LDS reads complete
    const h16* kbase  = K  + ((size_t)b * S_LEN + kb * 64) * DIM + hh * HEAD;
    const h16* vtbase = Vt + (size_t)(hh * HEAD) * BS + (size_t)b * S_LEN + kb * 64;
#if HAVE_TDM
    if (wave == 0) {
      tdm_load_tile_64x64((unsigned int)(uintptr_t)(void*)ldsK,  kbase,  DIM);
      tdm_load_tile_64x64((unsigned int)(uintptr_t)(void*)ldsVt, vtbase, BS);
      __builtin_amdgcn_s_wait_tensorcnt(0);
    }
#else
    for (int i = threadIdx.x; i < 64 * 64; i += 128) {
      const int kk = i >> 6, dd = i & 63;
      ldsK[kk * 64 + dd]  = kbase[(size_t)kk * DIM + dd];
      ldsVt[dd * 64 + kk] = vtbase[(size_t)dd * BS + kk];
    }
#endif
    __syncthreads();

    // --- scores = Q K^T, scaled by causal decay, staged as f16 ---
#pragma unroll
    for (int t = 0; t < 4; ++t) {
      v16h b0 = load_frag(ldsK + (t * 16) * 64 + 0,  64);
      v16h b1 = load_frag(ldsK + (t * 16) * 64 + 32, 64);
      v8f sc = {};
      sc = wmma_f16(aq0, b0, sc);
      sc = wmma_f16(aq1, b1, sc);
      const int m_glob = kb * 64 + t * 16 + (lane & 15);
#pragma unroll
      for (int r = 0; r < 8; ++r) {
        const int m_row  = r + ((lane >> 4) << 3);
        const int n_glob = row0 + m_row;
        const int d      = n_glob - m_glob;
        const float w = (d >= 0) ? exp2f(log2g * (float)d) : 0.0f;
        ldsS[(wave * 16 + m_row) * 64 + t * 16 + (lane & 15)] =
            (h16)(sc[r] * w);
      }
    }

    // --- Yacc += S(16x64) * V(64x64) --- (own wave's rows)
    const v16h as0 = load_frag(ldsS + (wave * 16) * 64 + 0,  64);
    const v16h as1 = load_frag(ldsS + (wave * 16) * 64 + 32, 64);
#pragma unroll
    for (int t = 0; t < 4; ++t) {
      v16h bv0 = load_frag(ldsVt + (t * 16) * 64 + 0,  64);
      v16h bv1 = load_frag(ldsVt + (t * 16) * 64 + 32, 64);
      acc[t] = wmma_f16(as0, bv0, acc[t]);
      acc[t] = wmma_f16(as1, bv1, acc[t]);
    }
  }

  // --- per-row GroupNorm over HEAD=64 (xor-shuffle within 16-lane half) ---
  float mean[8], rstd[8];
#pragma unroll
  for (int r = 0; r < 8; ++r) {
    float s1 = acc[0][r] + acc[1][r] + acc[2][r] + acc[3][r];
    float s2 = acc[0][r] * acc[0][r] + acc[1][r] * acc[1][r] +
               acc[2][r] * acc[2][r] + acc[3][r] * acc[3][r];
#pragma unroll
    for (int m = 1; m <= 8; m <<= 1) {
      s1 += __shfl_xor(s1, m, 32);
      s2 += __shfl_xor(s2, m, 32);
    }
    const float mu  = s1 * (1.0f / 64.0f);
    const float var = s2 * (1.0f / 64.0f) - mu * mu;
    mean[r] = mu;
    rstd[r] = rsqrtf(var + 1e-5f);
  }

  // --- GroupNorm affine + swish(g) gate, emit gated f16 P[B,S,DIM] ---
#pragma unroll
  for (int t = 0; t < 4; ++t) {
    const int dglob = hh * HEAD + t * 16 + (lane & 15);
    const float gw  = gn_w[dglob];
    const float gbv = gn_b[dglob];
#pragma unroll
    for (int r = 0; r < 8; ++r) {
      const int m_row  = r + ((lane >> 4) << 3);
      const int s_glob = row0 + m_row;
      const float yv = (acc[t][r] - mean[r]) * rstd[r] * gw + gbv;
      const size_t gidx = ((size_t)b * S_LEN + s_glob) * DIM + dglob;
      const float gv = (float)G[gidx];
      const float gate = gv / (1.0f + expf(-gv));   // swish
      P[gidx] = (h16)(gate * yv);
    }
  }
}

// ---------------------------------------------------------------------------
extern "C" void kernel_launch(void* const* d_in, const int* in_sizes, int n_in,
                              void* d_out, int out_size, void* d_ws, size_t ws_size,
                              hipStream_t stream) {
  (void)in_sizes; (void)n_in; (void)out_size; (void)ws_size;
  const float* x     = (const float*)d_in[0];
  const float* y     = (const float*)d_in[1];
  const float* ln1_w = (const float*)d_in[2];
  const float* ln1_b = (const float*)d_in[3];
  const float* W_Q   = (const float*)d_in[4];
  const float* W_K   = (const float*)d_in[5];
  const float* W_V   = (const float*)d_in[6];
  const float* gn_w  = (const float*)d_in[7];
  const float* gn_b  = (const float*)d_in[8];
  const float* W_G   = (const float*)d_in[9];
  const float* W_O   = (const float*)d_in[10];
  const float* ln2_w = (const float*)d_in[11];
  const float* ln2_b = (const float*)d_in[12];
  const float* ff_w1 = (const float*)d_in[13];
  const float* ff_b1 = (const float*)d_in[14];
  const float* ff_w2 = (const float*)d_in[15];
  const float* ff_b2 = (const float*)d_in[16];
  float* out = (float*)d_out;

  char* ws = (char*)d_ws;
  const size_t HB = (size_t)BS * DIM * sizeof(h16);  // 8 MB per f16 activation
  h16* xn  = (h16*)(ws + 0 * HB);   // dead after Q/G gemms
  h16* zn  = (h16*)(ws + 1 * HB);   // dead after K/V gemms
  h16* qb  = (h16*)(ws + 2 * HB);   // dead after retention
  h16* kbf = (h16*)(ws + 3 * HB);
  h16* vtb = (h16*)(ws + 4 * HB);   // V^T, [DIM][BS]
  h16* gb  = (h16*)(ws + 5 * HB);
  h16* pb  = (h16*)(ws + 6 * HB);   // dead after W_O gemm
  float* x1 = (float*)(ws + 0 * HB);  // f32, 2*HB, reuses xn+zn
  h16* x1n = (h16*)(ws + 2 * HB);     // reuses qb
  h16* hb  = (h16*)(ws + 3 * HB);     // f16 [BS,MLP], 2*HB, reuses kbf+vtb
  char* wsw = ws + 7 * HB;
  const size_t WB = (size_t)DIM * DIM * sizeof(h16);  // 128 KB
  h16* wqT = (h16*)(wsw + 0 * WB);
  h16* wkT = (h16*)(wsw + 1 * WB);
  h16* wvT = (h16*)(wsw + 2 * WB);
  h16* wgT = (h16*)(wsw + 3 * WB);
  h16* woT = (h16*)(wsw + 4 * WB);
  h16* w1T = (h16*)(wsw + 5 * WB);  // [MLP, DIM] -> 2*WB
  h16* w2T = (h16*)(wsw + 7 * WB);  // [DIM, MLP] -> 2*WB

  // 1) weight convert/transpose to f16 (B operand stored [N,K])
  transpose_cvt_kernel<<<(HEADS * DIM * HEAD + 255) / 256, 256, 0, stream>>>(W_Q, wqT, DIM, HEAD, HEADS);
  transpose_cvt_kernel<<<(HEADS * DIM * HEAD + 255) / 256, 256, 0, stream>>>(W_K, wkT, DIM, HEAD, HEADS);
  transpose_cvt_kernel<<<(HEADS * DIM * HEAD + 255) / 256, 256, 0, stream>>>(W_V, wvT, DIM, HEAD, HEADS);
  transpose_cvt_kernel<<<(DIM * DIM + 255) / 256, 256, 0, stream>>>(W_G, wgT, DIM, DIM, 1);
  transpose_cvt_kernel<<<(DIM * DIM + 255) / 256, 256, 0, stream>>>(W_O, woT, DIM, DIM, 1);
  transpose_cvt_kernel<<<(DIM * MLP + 255) / 256, 256, 0, stream>>>(ff_w1, w1T, DIM, MLP, 1);
  transpose_cvt_kernel<<<(MLP * DIM + 255) / 256, 256, 0, stream>>>(ff_w2, w2T, MLP, DIM, 1);

  // 2) PreNorm both streams
  ln_single<<<BS, 256, 0, stream>>>(x, ln1_w, ln1_b, xn);
  ln_single<<<BS, 256, 0, stream>>>(y, ln1_w, ln1_b, zn);

  // 3) Q/K/V/G projections (heads concatenated along N); V stored transposed
  dim3 g256(BS / 64, DIM / 64);
  gemm_kernel<<<g256, 128, 0, stream>>>(xn, wqT, nullptr, nullptr, qb,  nullptr, nullptr, BS, DIM, DIM, 0);
  gemm_kernel<<<g256, 128, 0, stream>>>(zn, wkT, nullptr, nullptr, kbf, nullptr, nullptr, BS, DIM, DIM, 0);
  gemm_kernel<<<g256, 128, 0, stream>>>(zn, wvT, nullptr, nullptr, nullptr, vtb, nullptr, BS, DIM, DIM, 0);
  gemm_kernel<<<g256, 128, 0, stream>>>(xn, wgT, nullptr, nullptr, gb,  nullptr, nullptr, BS, DIM, DIM, 0);

  // 4) fused retention + GroupNorm + swish gate (TDM-staged tiles)
  retention_kernel<<<dim3(S_LEN / 64, B_SZ * HEADS), 128, 0, stream>>>(
      qb, kbf, vtb, gb, gn_w, gn_b, pb);

  // 5) output projection + residual -> x1 (f32)
  gemm_kernel<<<g256, 128, 0, stream>>>(pb, woT, nullptr, x, nullptr, nullptr, x1, BS, DIM, DIM, 0);

  // 6) PreNorm1 -> FFN -> residual
  ln_single<<<BS, 256, 0, stream>>>(x1, ln2_w, ln2_b, x1n);
  gemm_kernel<<<dim3(BS / 64, MLP / 64), 128, 0, stream>>>(
      x1n, w1T, ff_b1, nullptr, hb, nullptr, nullptr, BS, MLP, DIM, 1 /*leaky*/);
  gemm_kernel<<<g256, 128, 0, stream>>>(
      hb, w2T, ff_b2, x1, nullptr, nullptr, out, BS, DIM, MLP, 0);
}
